// Grapher_45767171506482
// MI455X (gfx1250) — matrix operand order repaired and checked by reading, not compiled, not run on hardware
//
#include <hip/hip_runtime.h>
#include <hip/hip_bf16.h>

typedef __attribute__((ext_vector_type(16))) __bf16 v16bf;
typedef __attribute__((ext_vector_type(8)))  float  v8f;
typedef unsigned short u16t;

union Frag { v16bf v; unsigned u[8]; };

__device__ __forceinline__ u16t f2bf(float f) {
    unsigned u = __builtin_bit_cast(unsigned, f);
    unsigned r = (u + 0x7FFFu + ((u >> 16) & 1u)) >> 16;
    return (u16t)r;
}

__device__ __forceinline__ float gelu_exact(float v) {
    return 0.5f * v * (1.0f + erff(v * 0.7071067811865476f));
}

// ---------------------------------------------------------------------------
// Generic bf16 WMMA GEMM: Out[b] = A (MxK row-major) * B[b] (KxN) [+bias][+add]
// Block: 256 threads (8 waves). Block tile 64(M) x 128(N). Wave tile 16x64.
// ---------------------------------------------------------------------------
__global__ void gemm_bf16(const u16t* __restrict__ A, const u16t* __restrict__ Bm,
                          float* __restrict__ Out, int M, int N, int Kt,
                          long bStrideB, long bStrideOut,
                          const float* __restrict__ bias,
                          const float* __restrict__ addsrc, long bStrideAdd) {
    const int batch = blockIdx.z;
    const u16t* Bp = Bm + (long)batch * bStrideB;
    float* Op = Out + (long)batch * bStrideOut;
    const float* Addp = addsrc ? (addsrc + (long)batch * bStrideAdd) : nullptr;

    const int n0 = blockIdx.x * 128;
    const int m0 = blockIdx.y * 64;
    const int tid = threadIdx.x;
    const int w = tid >> 5, lane = tid & 31;
    const int wm = (w & 3) * 16, wn = (w >> 2) * 64;
    const int hx = lane >> 4, lm = lane & 15;

    __shared__ u16t As[64][34];   // row = m, col = k (pairs contiguous)
    __shared__ u16t Bs[128][34];  // row = n, col = k (transposed on load)

    v8f acc[4] = {};

    for (int kb = 0; kb < Kt; kb += 32) {
        __syncthreads();
        // A: 64 rows x 16 u32 (pairs of bf16), coalesced row-major copy
        for (int li = tid; li < 64 * 16; li += 256) {
            int r = li >> 4, c2 = li & 15;
            ((unsigned*)&As[r][0])[c2] =
                ((const unsigned*)(A + (long)(m0 + r) * Kt + kb))[c2];
        }
        // B: read rows k,k+1 (n contiguous -> coalesced), pack into Bs[n][k]
        for (int li = tid; li < 16 * 128; li += 256) {
            int k2 = li >> 7, n = li & 127;
            int k = k2 * 2;
            unsigned lo = Bp[(long)(kb + k) * N + n0 + n];
            unsigned hi = Bp[(long)(kb + k + 1) * N + n0 + n];
            *(unsigned*)&Bs[n][k] = lo | (hi << 16);
        }
        __syncthreads();

        Frag af;
#pragma unroll
        for (int p = 0; p < 8; ++p) {
            int K = ((p >> 2) << 4) + (hx << 3) + ((p & 3) << 1);
            af.u[p] = *(const unsigned*)&As[wm + lm][K];
        }
#pragma unroll
        for (int t = 0; t < 4; ++t) {
            Frag bf;
#pragma unroll
            for (int p = 0; p < 8; ++p) {
                int K = ((p >> 2) << 4) + (hx << 3) + ((p & 3) << 1);
                bf.u[p] = *(const unsigned*)&Bs[wn + t * 16 + lm][K];
            }
            acc[t] = __builtin_amdgcn_wmma_f32_16x16x32_bf16(
                false, af.v, false, bf.v, (short)0, acc[t], false, false);
        }
    }

#pragma unroll
    for (int t = 0; t < 4; ++t) {
        int col = n0 + wn + t * 16 + lm;
#pragma unroll
        for (int r = 0; r < 8; ++r) {
            int row = m0 + wm + hx * 8 + r;
            float v = acc[t][r];
            if (bias) v += bias[row];
            long o = (long)row * N + col;
            if (Addp) v += Addp[o];
            Op[o] = v;
        }
    }
}

// ---------------------------------------------------------------------------
// Per-channel BatchNorm stats over (B,N): scale = g*rsqrt(var+eps),
// shift = b - mean*scale.   One block per channel.
// ---------------------------------------------------------------------------
__global__ void bn_stats(const float* __restrict__ t, const float* __restrict__ gamma,
                         const float* __restrict__ beta,
                         float* __restrict__ scale, float* __restrict__ shift) {
    const int c = blockIdx.x;
    float s = 0.f, s2 = 0.f;
    for (int i = threadIdx.x; i < 4 * 4096; i += 256) {
        int b = i >> 12, n = i & 4095;
        float v = t[((long)b * 192 + c) * 4096 + n];
        s += v; s2 += v * v;
    }
    __shared__ float rs[256], rq[256];
    rs[threadIdx.x] = s; rq[threadIdx.x] = s2;
    __syncthreads();
    for (int off = 128; off > 0; off >>= 1) {
        if (threadIdx.x < off) { rs[threadIdx.x] += rs[threadIdx.x + off];
                                 rq[threadIdx.x] += rq[threadIdx.x + off]; }
        __syncthreads();
    }
    if (threadIdx.x == 0) {
        float mean = rs[0] * (1.f / 16384.f);
        float var  = rq[0] * (1.f / 16384.f) - mean * mean;
        float sc = gamma[c] * rsqrtf(var + 1e-5f);
        scale[c] = sc;
        shift[c] = beta[c] - mean * sc;
    }
}

// ---------------------------------------------------------------------------
// Apply BN0 and transpose (B,C,N)->(B,N,C), write fp32 + bf16 copies.
// grid (N/64, C/32, B), 256 threads, LDS transpose tile.
// ---------------------------------------------------------------------------
__global__ void bn0_apply(const float* __restrict__ t, const float* __restrict__ scale,
                          const float* __restrict__ shift,
                          float* __restrict__ hncf, u16t* __restrict__ hncb) {
    const int b = blockIdx.z, c0 = blockIdx.y * 32, n0 = blockIdx.x * 64;
    __shared__ float S[32][65];
    const int tid = threadIdx.x;
    for (int li = tid; li < 2048; li += 256) {
        int r = li >> 6, col = li & 63;
        float v = t[((long)b * 192 + c0 + r) * 4096 + n0 + col];
        S[r][col] = v * scale[c0 + r] + shift[c0 + r];
    }
    __syncthreads();
    for (int li = tid; li < 2048; li += 256) {
        int n = li >> 5, c = li & 31;
        float v = S[c][n];
        long o = ((long)b * 4096 + n0 + n) * 192 + c0 + c;
        hncf[o] = v;
        hncb[o] = f2bf(v);
    }
}

// sq[b,n] = sum_c h^2 : one wave per point, shuffle reduce (deterministic)
__global__ void sq_kernel(const float* __restrict__ hncf, float* __restrict__ sq) {
    const int w = threadIdx.x >> 5, lane = threadIdx.x & 31;
    const long pt = (long)blockIdx.x * 8 + w;
    const float* r = hncf + pt * 192;
    float s = 0.f;
#pragma unroll
    for (int g = 0; g < 6; ++g) { float v = r[lane + 32 * g]; s += v * v; }
    for (int off = 16; off > 0; off >>= 1) s += __shfl_down(s, off, 32);
    if (lane == 0) sq[pt] = s;
}

// ---------------------------------------------------------------------------
// Fused KNN: Gram via WMMA + streaming top-16 (smallest dist) per row.
// Block: 128 threads (4 waves). i-block = 64 rows, j streamed in chunks of 64.
// dist = sq_i + sq_j - 2*<f_i,f_j>. LDS < 64KB via Bt/Dst union.
// ---------------------------------------------------------------------------
union BtDst { u16t Bt[64][194]; float Dst[4][16][65]; };

__global__ void knn_kernel(const u16t* __restrict__ hncb, const float* __restrict__ sq,
                           int* __restrict__ idxout) {
    const int NB = 4096, CC = 192;
    const int b = blockIdx.y;
    const int i0 = blockIdx.x * 64;
    const int tid = threadIdx.x, w = tid >> 5, lane = tid & 31;
    const int hx = lane >> 4, lm = lane & 15;

    __shared__ u16t  Af[64][194];
    __shared__ float Td[4][32][16];
    __shared__ u16t  Ti[4][32][16];
    __shared__ float sqj_s[64];
    __shared__ BtDst ub;

    // load A block (64 rows x 192 c) once, as u32 pairs
    for (int li = tid; li < 64 * 96; li += 128) {
        int r = li / 96, c2 = li % 96;
        ((unsigned*)&Af[r][0])[c2] =
            ((const unsigned*)(hncb + ((long)b * NB + i0 + r) * CC))[c2];
    }
    // per-lane row norms (rows this lane's accumulators cover)
    float sqi[8];
#pragma unroll
    for (int r = 0; r < 8; ++r)
        sqi[r] = sq[(long)b * NB + i0 + w * 16 + hx * 8 + r];
    // init top-k lists
#pragma unroll
    for (int e = 0; e < 16; ++e) { Td[w][lane][e] = __builtin_inff(); Ti[w][lane][e] = 0; }
    float curMax = __builtin_inff();
    int   curPos = 0;
    __syncthreads();

    // hoist all 6 A fragments (invariant over j)
    Frag afr[6];
#pragma unroll
    for (int ks = 0; ks < 6; ++ks)
#pragma unroll
        for (int p = 0; p < 8; ++p) {
            int K = ks * 32 + ((p >> 2) << 4) + (hx << 3) + ((p & 3) << 1);
            afr[ks].u[p] = *(const unsigned*)&Af[w * 16 + lm][K];
        }

    const int selRow = lane >> 1, selHalf = lane & 1;

    for (int jc = 0; jc < NB; jc += 64) {
        __syncthreads();  // previous Dst use finished
        for (int li = tid; li < 64 * 96; li += 128) {
            int r = li / 96, c2 = li % 96;
            ((unsigned*)&ub.Bt[r][0])[c2] =
                ((const unsigned*)(hncb + ((long)b * NB + jc + r) * CC))[c2];
        }
        if (tid < 64) sqj_s[tid] = sq[(long)b * NB + jc + tid];
        __syncthreads();

        v8f acc[4] = {};
#pragma unroll
        for (int t = 0; t < 4; ++t) {
#pragma unroll
            for (int ks = 0; ks < 6; ++ks) {
                Frag bf;
#pragma unroll
                for (int p = 0; p < 8; ++p) {
                    int K = ks * 32 + ((p >> 2) << 4) + (hx << 3) + ((p & 3) << 1);
                    bf.u[p] = *(const unsigned*)&ub.Bt[t * 16 + lm][K];
                }
                acc[t] = __builtin_amdgcn_wmma_f32_16x16x32_bf16(
                    false, afr[ks].v, false, bf.v, (short)0, acc[t], false, false);
            }
        }
        __syncthreads();  // everyone done reading Bt before Dst overwrite
#pragma unroll
        for (int t = 0; t < 4; ++t)
#pragma unroll
            for (int r = 0; r < 8; ++r) {
                float d = sqi[r] + sqj_s[t * 16 + lm] - 2.0f * acc[t][r];
                ub.Dst[w][hx * 8 + r][t * 16 + lm] = d;
            }
        __syncthreads();

        // streaming selection: 2 lanes per row, each scans 32 candidates
        for (int s = 0; s < 32; ++s) {
            int col = selHalf * 32 + s;
            float d = ub.Dst[w][selRow][col];
            if (d < curMax) {
                Td[w][lane][curPos] = d;
                Ti[w][lane][curPos] = (u16t)(jc + col);
                float mx = -__builtin_inff(); int mp = 0;
#pragma unroll
                for (int e = 0; e < 16; ++e) {
                    float v = Td[w][lane][e];
                    if (v > mx) { mx = v; mp = e; }
                }
                curMax = mx; curPos = mp;
            }
        }
    }
    __syncthreads();

    // merge the two 16-lists per row -> final 16 (order irrelevant: max-agg)
    if (lane < 16) {
        int m = lane;
        for (int t = 0; t < 16; ++t) {
            float best = __builtin_inff(); int bl = 0, be = 0;
            for (int e = 0; e < 16; ++e) {
                float v0 = Td[w][2 * m][e];
                if (v0 < best) { best = v0; bl = 0; be = e; }
                float v1 = Td[w][2 * m + 1][e];
                if (v1 < best) { best = v1; bl = 1; be = e; }
            }
            idxout[((long)b * NB + i0 + w * 16 + m) * 16 + t] = (int)Ti[w][2 * m + bl][be];
            Td[w][2 * m + bl][be] = __builtin_inff();
        }
    }
}

// ---------------------------------------------------------------------------
// MRConv build: H2[b,2c,n]=feat, H2[b,2c+1,n]=max_j(h_j - h_i), bf16 output.
// One wave per point; lane covers 6 channels (c = lane + 32s).
// ---------------------------------------------------------------------------
__global__ void mr_build(const float* __restrict__ hncf, const int* __restrict__ idx,
                         u16t* __restrict__ H2) {
    const int w = threadIdx.x >> 5, lane = threadIdx.x & 31;
    const long pt = (long)blockIdx.x * 8 + w;
    const int b = (int)(pt >> 12), n = (int)(pt & 4095);
    const float* base = hncf + ((long)b * 4096 + n) * 192;
    float fi[6], agg[6];
#pragma unroll
    for (int s = 0; s < 6; ++s) { fi[s] = base[lane + 32 * s]; agg[s] = -__builtin_inff(); }
    const int* ip = idx + pt * 16;
    for (int k = 0; k < 16; ++k) {
        int j = ip[k];
        const float* jb = hncf + ((long)b * 4096 + j) * 192;
#pragma unroll
        for (int s = 0; s < 6; ++s) {
            float df = jb[lane + 32 * s] - fi[s];
            agg[s] = fmaxf(agg[s], df);
        }
    }
    u16t* hb = H2 + (long)b * 384 * 4096;
#pragma unroll
    for (int s = 0; s < 6; ++s) {
        int c = lane + 32 * s;
        hb[(long)(2 * c) * 4096 + n]     = f2bf(fi[s]);
        hb[(long)(2 * c + 1) * 4096 + n] = f2bf(agg[s]);
    }
}

// BN1 apply + gelu(gelu(.)), write fp32 (for stats) + bf16 (for fc2 GEMM)
__global__ void bn_gelu2(const float* __restrict__ t, const float* __restrict__ scale,
                         const float* __restrict__ shift,
                         float* __restrict__ uf, u16t* __restrict__ ubf, long total) {
    long i = (long)blockIdx.x * 256 + threadIdx.x;
    if (i >= total) return;
    int c = (int)((i >> 12) % 192);
    float v = t[i] * scale[c] + shift[c];
    v = gelu_exact(gelu_exact(v));
    uf[i] = v;
    ubf[i] = f2bf(v);
}

// Fold final BN into fc2: w2p[o,c]=bf16(w2*scale2[c]); bias2[o]=sum_c w2*shift2[c]
__global__ void fold_w2(const float* __restrict__ w2, const float* __restrict__ scale2,
                        const float* __restrict__ shift2,
                        u16t* __restrict__ w2p, float* __restrict__ bias2) {
    const int o = blockIdx.x;
    float s = 0.f;
    for (int c = threadIdx.x; c < 192; c += 64) {
        float wv = w2[o * 192 + c];
        w2p[o * 192 + c] = f2bf(wv * scale2[c]);
        s += wv * shift2[c];
    }
    __shared__ float red[64];
    red[threadIdx.x] = s; __syncthreads();
    for (int off = 32; off > 0; off >>= 1) {
        if (threadIdx.x < off) red[threadIdx.x] += red[threadIdx.x + off];
        __syncthreads();
    }
    if (threadIdx.x == 0) bias2[o] = red[0];
}

__global__ void cvt_bf16(const float* __restrict__ src, u16t* __restrict__ dst, long n) {
    long i = (long)blockIdx.x * 256 + threadIdx.x;
    if (i < n) dst[i] = f2bf(src[i]);
}

// ---------------------------------------------------------------------------
extern "C" void kernel_launch(void* const* d_in, const int* in_sizes, int n_in,
                              void* d_out, int out_size, void* d_ws, size_t ws_size,
                              hipStream_t stream) {
    (void)in_sizes; (void)n_in; (void)out_size; (void)ws_size;
    const float* x    = (const float*)d_in[0];
    const float* wfc1 = (const float*)d_in[1];
    const float* wfc2 = (const float*)d_in[2];
    const float* wmr  = (const float*)d_in[3];
    const float* bmr  = (const float*)d_in[4];
    const float* g0   = (const float*)d_in[5];
    const float* b0   = (const float*)d_in[6];
    const float* gm   = (const float*)d_in[7];
    const float* bm   = (const float*)d_in[8];
    const float* g1   = (const float*)d_in[9];
    const float* b1   = (const float*)d_in[10];
    float* out = (float*)d_out;

    const long Bn = 4, Cc = 192, Nn = 4096;
    const long BCN = Bn * Cc * Nn;

    char* ws = (char*)d_ws;
    size_t off = 0;
    auto alloc = [&](size_t bytes) { size_t o = off; off = (off + bytes + 255) & ~(size_t)255; return o; };
    float* t     = (float*)(ws + alloc(BCN * 4));            // fc1 out, reused for mr out
    float* hncf  = (float*)(ws + alloc(BCN * 4));            // h (B,N,C) fp32
    u16t*  hncb  = (u16t*) (ws + alloc(BCN * 2));            // h (B,N,C) bf16
    u16t*  xbf   = (u16t*) (ws + alloc(BCN * 2));            // x bf16, reused for u bf16
    char*  h2u   = ws + alloc(Bn * 2 * Cc * Nn * 2);         // H2 bf16, reused for u fp32
    u16t*  H2    = (u16t*)h2u;
    float* uf    = (float*)h2u;
    float* sq    = (float*)(ws + alloc(Bn * Nn * 4));
    int*   idxb  = (int*)  (ws + alloc(Bn * Nn * 16 * 4));
    u16t*  wA1   = (u16t*) (ws + alloc(192 * 192 * 2));
    u16t*  wAmr  = (u16t*) (ws + alloc(192 * 384 * 2));
    u16t*  w2p   = (u16t*) (ws + alloc(192 * 192 * 2));
    float* scale0 = (float*)(ws + alloc(192 * 4));
    float* shift0 = (float*)(ws + alloc(192 * 4));
    float* scale1 = (float*)(ws + alloc(192 * 4));
    float* shift1 = (float*)(ws + alloc(192 * 4));
    float* scale2 = (float*)(ws + alloc(192 * 4));
    float* shift2 = (float*)(ws + alloc(192 * 4));
    float* bias2  = (float*)(ws + alloc(192 * 4));

    // 1) bf16 conversions
    cvt_bf16<<<(192 * 192 + 255) / 256, 256, 0, stream>>>(wfc1, wA1, 192 * 192);
    cvt_bf16<<<(192 * 384 + 255) / 256, 256, 0, stream>>>(wmr, wAmr, 192 * 384);
    cvt_bf16<<<(int)((BCN + 255) / 256), 256, 0, stream>>>(x, xbf, BCN);

    // 2) fc1 GEMM (WMMA): t = W1 * x   per batch
    gemm_bf16<<<dim3(32, 3, 4), 256, 0, stream>>>(wA1, xbf, t, 192, 4096, 192,
        Cc * Nn, Cc * Nn, nullptr, nullptr, 0);

    // 3) BN0 stats + apply (transpose to (B,N,C)) + row norms
    bn_stats<<<192, 256, 0, stream>>>(t, g0, b0, scale0, shift0);
    bn0_apply<<<dim3(64, 6, 4), 256, 0, stream>>>(t, scale0, shift0, hncf, hncb);
    sq_kernel<<<2048, 256, 0, stream>>>(hncf, sq);

    // 4) fused KNN (Gram via WMMA + streaming top-16)
    knn_kernel<<<dim3(64, 4), 128, 0, stream>>>(hncb, sq, idxb);

    // 5) MRConv: gather/max-agg/interleave -> H2 (bf16), then GEMM + bias
    mr_build<<<2048, 256, 0, stream>>>(hncf, idxb, H2);
    gemm_bf16<<<dim3(32, 3, 4), 256, 0, stream>>>(wAmr, H2, t, 192, 4096, 384,
        2 * Cc * Nn, Cc * Nn, bmr, nullptr, 0);

    // 6) BN1 + gelu(gelu(.)) -> u (fp32 for stats, bf16 for fc2)
    bn_stats<<<192, 256, 0, stream>>>(t, gm, bm, scale1, shift1);
    bn_gelu2<<<(int)((BCN + 255) / 256), 256, 0, stream>>>(t, scale1, shift1, uf, xbf, BCN);

    // 7) BN2 folded into fc2 weights; fc2 GEMM + bias + residual -> d_out
    bn_stats<<<192, 256, 0, stream>>>(uf, g1, b1, scale2, shift2);
    fold_w2<<<192, 64, 0, stream>>>(wfc2, scale2, shift2, w2p, bias2);
    gemm_bf16<<<dim3(32, 3, 4), 256, 0, stream>>>(w2p, xbf, out, 192, 4096, 192,
        Cc * Nn, Cc * Nn, bias2, x, Cc * Nn);
}
